// ImprovedDiPPeRModel_26792005992650
// MI455X (gfx1250) — matrix operand order repaired and checked by reading, not compiled, not run on hardware
//
#include <hip/hip_runtime.h>
#include <hip/hip_bf16.h>
#include <cstdint>
#include <cstddef>

typedef _Float16 h16;
typedef __attribute__((ext_vector_type(16))) _Float16 v16h;
typedef __attribute__((ext_vector_type(8)))  _Float16 v8h;
typedef __attribute__((ext_vector_type(8)))  float    v8f;
typedef __attribute__((ext_vector_type(4)))  unsigned u32x4;
typedef __attribute__((ext_vector_type(4)))  int      i32x4;
typedef __attribute__((ext_vector_type(8)))  int      i32x8;

#define EPS_BN 1e-5f
#define LDK 72   // LDS row stride in halfs (64 data + 8 pad -> 144B, conflict-free)

// ---- Tensor Data Mover availability (device pass on capable toolchains) ----
#if defined(__AMDGCN__) && defined(__has_builtin)
#if __has_builtin(__builtin_amdgcn_tensor_load_to_lds)
#define USE_TDM 1
#endif
#endif
#ifndef USE_TDM
#define USE_TDM 0
#endif

#if USE_TDM
// Issue one TDM 2D tile load: global (rows x tile0 elems, rowStride elems, 2B
// elements) -> LDS at ldsByte with +8-half pad per 64-half row (matches LDK).
// dim0Remain/dim1Remain are the tensor extents from the tile start: TDM
// zero-fills out-of-bounds reads, which implements the K-tail padding.
__device__ inline void tdm_load_2d(unsigned ldsByte, const void* gaddr,
                                   unsigned dim0Remain, unsigned dim1Remain,
                                   unsigned tile0, unsigned tile1,
                                   unsigned rowStrideElems) {
  unsigned long long ga = (unsigned long long)(uintptr_t)gaddr;
  u32x4 g0;
  g0[0] = 1u;                                            // count=1, user D#
  g0[1] = ldsByte;                                       // lds_addr [63:32]
  g0[2] = (unsigned)(ga & 0xFFFFFFFFu);                  // global_addr lo
  g0[3] = ((unsigned)(ga >> 32) & 0x01FFFFFFu) | (2u << 30);  // addr hi | type=2
  i32x8 g1;
  // data_size=1 (2B) | pad_enable | pad_interval=4 (32 DW) | pad_amount=3 (4 DW)
  g1[0] = (int)((1u << 16) | (1u << 20) | (4u << 22) | (3u << 25));
  g1[1] = (int)((dim0Remain & 0xFFFFu) << 16);           // tensor_dim0 [79:48]
  g1[2] = (int)((dim0Remain >> 16) | ((dim1Remain & 0xFFFFu) << 16));
  g1[3] = (int)((dim1Remain >> 16) | (tile0 << 16));     // tile_dim0 [127:112]
  g1[4] = (int)(tile1 & 0xFFFFu);                        // tile_dim1 [143:128]
  g1[5] = (int)rowStrideElems;                           // tensor_dim0_stride lo
  g1[6] = 0;
  g1[7] = 0;
  i32x4 z4 = {0, 0, 0, 0};
#if __clang_major__ >= 23
  i32x8 z8 = {0, 0, 0, 0, 0, 0, 0, 0};
  __builtin_amdgcn_tensor_load_to_lds(g0, g1, z4, z4, z8, 0);
#else
  __builtin_amdgcn_tensor_load_to_lds(g0, g1, z4, z4, 0);
#endif
}
#endif  // USE_TDM

// ---------------------------------------------------------------------------
// Weight / parameter prep kernels (run every call; cheap, deterministic)
// ---------------------------------------------------------------------------

// OIHW f32 -> [O][tap][I] f16  (tap = ky*3+kx) : makes WMMA B-fragments contiguous
__global__ __launch_bounds__(256) void k_prep_conv_w(const float* __restrict__ src,
                                                     h16* __restrict__ dst, int O, int I) {
  int n = O * 9 * I;
  int t = blockIdx.x * 256 + threadIdx.x;
  if (t >= n) return;
  int o   = t / (9 * I);
  int r   = t - o * 9 * I;
  int tap = r / I;
  int ci  = r - tap * I;
  dst[t] = (h16)src[(o * I + ci) * 9 + tap];
}

// [N][K] f32 -> [Np][Kp] f16 zero-padded
__global__ __launch_bounds__(256) void k_prep_fc_w(const float* __restrict__ src,
                                                   h16* __restrict__ dst,
                                                   int N, int K, int Np, int Kp) {
  long long n = (long long)Np * Kp;
  long long t = (long long)blockIdx.x * 256 + threadIdx.x;
  if (t >= n) return;
  int nn = (int)(t / Kp);
  int kk = (int)(t - (long long)nn * Kp);
  float v = (nn < N && kk < K) ? src[(long long)nn * K + kk] : 0.f;
  dst[t] = (h16)v;
}

// Fold conv/FC bias + BN into scale/bias:  y = relu(acc*scale + bias)
__global__ __launch_bounds__(256) void k_prep_sb(const float* __restrict__ b,
                                                 const float* __restrict__ g,
                                                 const float* __restrict__ be,
                                                 const float* __restrict__ m,
                                                 const float* __restrict__ v,
                                                 float* __restrict__ sc,
                                                 float* __restrict__ bi, int n) {
  int t = blockIdx.x * 256 + threadIdx.x;
  if (t >= n) return;
  float s = g[t] * rsqrtf(v[t] + EPS_BN);
  sc[t] = s;
  bi[t] = (b[t] - m[t]) * s + be[t];
}

__global__ __launch_bounds__(256) void k_prep_sb_final(const float* __restrict__ b,
                                                       float* __restrict__ sc,
                                                       float* __restrict__ bi,
                                                       int nValid, int nTot) {
  int t = blockIdx.x * 256 + threadIdx.x;
  if (t >= nTot) return;
  sc[t] = 1.f;
  bi[t] = (t < nValid) ? b[t] : 0.f;
}

// ---------------------------------------------------------------------------
// Input assembly: x0[b][y][x][c] = {cost, heat(start), heat(goal), 0}  (NHWC f16)
// ---------------------------------------------------------------------------
__global__ __launch_bounds__(256) void k_build_x0(const float* __restrict__ cost,
                                                  const float* __restrict__ sp,
                                                  const float* __restrict__ gp,
                                                  h16* __restrict__ x0) {
  int t = blockIdx.x * 256 + threadIdx.x;       // over B*120*120
  if (t >= 256 * 120 * 120) return;
  int b = t / 14400;
  int r = t - b * 14400;
  int y = r / 120;
  int x = r - y * 120;
  float sx = fminf(fmaxf(floorf((sp[b*2+0] + 6.f) * (119.f/12.f)), 0.f), 119.f);
  float sy = fminf(fmaxf(floorf((sp[b*2+1] + 6.f) * (119.f/12.f)), 0.f), 119.f);
  float gx = fminf(fmaxf(floorf((gp[b*2+0] + 6.f) * (119.f/12.f)), 0.f), 119.f);
  float gy = fminf(fmaxf(floorf((gp[b*2+1] + 6.f) * (119.f/12.f)), 0.f), 119.f);
  float fy = (float)y, fx = (float)x;
  float hs = expf(-((fy - sy)*(fy - sy) + (fx - sx)*(fx - sx)) * 0.125f);
  float hg = expf(-((fy - gy)*(fy - gy) + (fx - gx)*(fx - gx)) * 0.125f);
  h16* o = x0 + (size_t)t * 4;
  o[0] = (h16)cost[t];
  o[1] = (h16)hs;
  o[2] = (h16)hg;
  o[3] = (h16)0.f;
}

// ---------------------------------------------------------------------------
// Conv layer 0 (Cin=3, ~2% of FLOPs): direct VALU conv, NHWC(C=4) -> NHWC(C=32)
// ---------------------------------------------------------------------------
__global__ __launch_bounds__(256) void k_conv1(const h16* __restrict__ x0,
                                               const float* __restrict__ w,  // OIHW [32][3][3][3]
                                               const float* __restrict__ sc,
                                               const float* __restrict__ bi,
                                               h16* __restrict__ out) {
  unsigned t = blockIdx.x * 256 + threadIdx.x;   // B*14400*32 = 117,964,800 < 2^31
  int co = (int)(t & 31);
  unsigned P = t >> 5;
  int b = (int)(P / 14400u);
  int r = (int)(P - (unsigned)b * 14400u);
  int y = r / 120;
  int x = r - y * 120;
  float acc = 0.f;
  for (int ky = 0; ky < 3; ++ky) {
    int iy = y + ky - 1; if (iy < 0 || iy >= 120) continue;
    for (int kx = 0; kx < 3; ++kx) {
      int ix = x + kx - 1; if (ix < 0 || ix >= 120) continue;
      const h16* p = x0 + (((size_t)b * 120 + iy) * 120 + ix) * 4;
      const float* wp = w + co * 27 + ky * 3 + kx;            // ci stride = 9
      acc += (float)p[0] * wp[0] + (float)p[1] * wp[9] + (float)p[2] * wp[18];
    }
  }
  float v = acc * sc[co] + bi[co];
  v = v > 0.f ? v : 0.f;
  out[(size_t)P * 32 + co] = (h16)v;
}

// ---------------------------------------------------------------------------
// WMMA implicit-GEMM conv (3x3 SAME), NHWC f16, Cin in {32,64,128} (pow2).
// Block = 256 thr = 8 waves; tile M=128 spatial x N=32 couts, K staged 64-wide.
// Activations gathered manually (halo logic); weight tile streamed by the
// Tensor Data Mover when available (TENSORcnt + s_wait_tensorcnt).
// ---------------------------------------------------------------------------
__global__ __launch_bounds__(256) void k_conv_wmma(const h16* __restrict__ in,
                                                   h16* __restrict__ out,
                                                   const h16* __restrict__ w,   // [Cout][9][Cin]
                                                   const float* __restrict__ sc,
                                                   const float* __restrict__ bi,
                                                   int B, int S, int Cin, int Cout,
                                                   int log2Cin) {
  __shared__ __align__(16) h16 As[128 * LDK];
  __shared__ __align__(16) h16 Bs[32 * LDK];
  const int tid  = threadIdx.x;
  const int lane = tid & 31;
  const int wave = tid >> 5;
  const unsigned SS = (unsigned)(S * S);
  const unsigned total = (unsigned)B * SS;
  const unsigned mBase = (unsigned)blockIdx.x * 128u;
  const int nBase = blockIdx.y * 32;
  const int Ktot = 9 * Cin;
  const int cmask = Cin - 1;

  // A tile: 128 rows x 64 k = 1024 x 8-half chunks; thread handles rows
  // r_i = (tid>>3) + 32*i (i=0..3) at column koffA = (tid&7)*8.
  const int koffA = (tid & 7) * 8;
  const int rA    = tid >> 3;                 // 0..31
  int pbA[4], pyA[4], pxA[4];
  bool pvA[4];
  #pragma unroll
  for (int i = 0; i < 4; ++i) {
    unsigned P = mBase + (unsigned)(rA + 32 * i);
    pvA[i] = (P < total);
    unsigned Pc = pvA[i] ? P : 0u;
    unsigned pb = Pc / SS;
    unsigned rem = Pc - pb * SS;
    unsigned py = rem / (unsigned)S;
    pbA[i] = (int)pb;
    pyA[i] = (int)py;
    pxA[i] = (int)(rem - py * (unsigned)S);
  }
#if !USE_TDM
  const int nB = tid >> 3;
  const h16* wRow = w + (size_t)(nBase + nB) * 9 * Cin;
#else
  const h16* wBase = w + (size_t)nBase * 9 * Cin;   // block's 32 weight rows
#endif

  v8f acc0 = {}, acc1 = {};
  const int nStages = (Ktot + 63) >> 6;
  for (int kb = 0; kb < nStages; ++kb) {
    const int ks = kb << 6;
    const int kglob = ks + koffA;               // this thread's staged K base
    const int tap = kglob >> log2Cin;           // Cin is pow2
    const int kin = kglob & cmask;
    const bool kvalid = (kglob < Ktot);
    const int dy = tap / 3 - 1;
    const int dx = tap - (tap / 3) * 3 - 1;
    // stage A (4 chunks, manual gather with halo/zero-fill)
    #pragma unroll
    for (int i = 0; i < 4; ++i) {
      const int iy = pyA[i] + dy, ix = pxA[i] + dx;
      const bool ok = kvalid && pvA[i] && iy >= 0 && iy < S && ix >= 0 && ix < S;
      v8h av = {};
      if (ok)
        av = *(const v8h*)(in + (((size_t)pbA[i] * S + iy) * S + ix) * Cin + kin);
      *(v8h*)(As + (rA + 32 * i) * LDK + koffA) = av;
    }
    // stage B (weight tile): TDM 2D descriptor or manual fallback
#if USE_TDM
    if (tid == 0) {
      tdm_load_2d((unsigned)(uintptr_t)Bs, wBase + ks,
                  (unsigned)(Ktot - ks), 32u, 64u, 32u, (unsigned)Ktot);
      __builtin_amdgcn_s_wait_tensorcnt((short)0);
    }
#else
    {
      v8h bvv = {};
      if (kvalid) bvv = *(const v8h*)(wRow + (size_t)tap * Cin + kin);
      *(v8h*)(Bs + nB * LDK + koffA) = bvv;
    }
#endif
    __syncthreads();
    // compute: 2 K-chunks x 2 N-subtiles = 4 WMMA per wave
    const int half = lane >> 4, l16 = lane & 15;
    const int kh = half * 8;
    const h16* ap = As + (wave * 16 + l16) * LDK;
    #pragma unroll
    for (int kc2 = 0; kc2 < 2; ++kc2) {
      const int ksl = kc2 * 32;
      v8h alo = *(const v8h*)(ap + ksl + kh);
      v8h ahi = *(const v8h*)(ap + ksl + 16 + kh);
      v16h a;
      #pragma unroll
      for (int i = 0; i < 8; ++i) { a[i] = alo[i]; a[8 + i] = ahi[i]; }
      #pragma unroll
      for (int ns = 0; ns < 2; ++ns) {
        const h16* bp = Bs + (ns * 16 + l16) * LDK + ksl + half * 8;
        v8h blo = *(const v8h*)(bp + 0);
        v8h bhi = *(const v8h*)(bp + 16);
        v16h bv;
        #pragma unroll
        for (int i = 0; i < 8; ++i) { bv[i] = blo[i]; bv[8 + i] = bhi[i]; }
        if (ns == 0)
          acc0 = __builtin_amdgcn_wmma_f32_16x16x32_f16(false, a, false, bv,
                                                        (short)0, acc0, false, false);
        else
          acc1 = __builtin_amdgcn_wmma_f32_16x16x32_f16(false, a, false, bv,
                                                        (short)0, acc1, false, false);
      }
    }
    __syncthreads();
  }
  // epilogue: fused BN + ReLU  (D: n = lane&15, m = 8*(lane>>4)+r)
  const int half = lane >> 4, l16 = lane & 15;
  #pragma unroll
  for (int ns = 0; ns < 2; ++ns) {
    const int n = nBase + ns * 16 + l16;
    const float s = sc[n], o = bi[n];
    #pragma unroll
    for (int r = 0; r < 8; ++r) {
      const unsigned Pm = mBase + (unsigned)(wave * 16 + half * 8 + r);
      if (Pm < total) {
        float v = (ns == 0 ? acc0[r] : acc1[r]) * s + o;
        v = v > 0.f ? v : 0.f;
        out[(size_t)Pm * Cout + n] = (h16)v;
      }
    }
  }
}

// ---------------------------------------------------------------------------
// 2x2 max pool, NHWC f16
// ---------------------------------------------------------------------------
__global__ __launch_bounds__(256) void k_maxpool(const h16* __restrict__ in,
                                                 h16* __restrict__ out,
                                                 int B, int S, int C) {
  int So = S >> 1;
  unsigned n = (unsigned)B * So * So * C;
  unsigned t = blockIdx.x * 256 + threadIdx.x;
  if (t >= n) return;
  int c = (int)(t % (unsigned)C);
  unsigned q = t / (unsigned)C;
  int x = (int)(q % (unsigned)So); q /= (unsigned)So;
  int y = (int)(q % (unsigned)So);
  int b = (int)(q / (unsigned)So);
  const h16* p = in + (((size_t)b * S + 2 * y) * S + 2 * x) * C + c;
  float a0 = (float)p[0], a1 = (float)p[C];
  float a2 = (float)p[(size_t)S * C], a3 = (float)p[(size_t)S * C + C];
  out[t] = (h16)fmaxf(fmaxf(a0, a1), fmaxf(a2, a3));
}

// ---------------------------------------------------------------------------
// AdaptiveAvgPool 15->8; writes h0 in NCHW-flatten order (c*64 + p*8 + q)
// ---------------------------------------------------------------------------
#define KP1 16448   // 16388 padded to multiple of 64
__global__ __launch_bounds__(256) void k_adpool(const h16* __restrict__ in,
                                                h16* __restrict__ h) {
  int t = blockIdx.x * 256 + threadIdx.x;       // B*256*64
  if (t >= 256 * 256 * 64) return;
  int q = t & 7;
  int p = (t >> 3) & 7;
  int c = (t >> 6) & 255;
  int b = t >> 14;
  int sy = (p * 15) >> 3, ey = ((p + 1) * 15 + 7) >> 3;
  int sx = (q * 15) >> 3, ex = ((q + 1) * 15 + 7) >> 3;
  float s = 0.f;
  for (int yy = sy; yy < ey; ++yy)
    for (int xx = sx; xx < ex; ++xx)
      s += (float)in[(((size_t)b * 15 + yy) * 15 + xx) * 256 + c];
  s /= (float)((ey - sy) * (ex - sx));
  h[(size_t)b * KP1 + c * 64 + p * 8 + q] = (h16)s;
}

// tail of h: [16384..16388) = start,goal; [16388..16448) = K-pad zeros
__global__ __launch_bounds__(256) void k_htail(const float* __restrict__ sp,
                                               const float* __restrict__ gp,
                                               h16* __restrict__ h) {
  int t = blockIdx.x * 256 + threadIdx.x;       // 256*64
  if (t >= 256 * 64) return;
  int b = t >> 6, j = t & 63;
  float v = 0.f;
  if      (j == 0) v = sp[b * 2 + 0];
  else if (j == 1) v = sp[b * 2 + 1];
  else if (j == 2) v = gp[b * 2 + 0];
  else if (j == 3) v = gp[b * 2 + 1];
  h[(size_t)b * KP1 + 16384 + j] = (h16)v;
}

// ---------------------------------------------------------------------------
// WMMA GEMM for FC: C[256][N] = A[256][Kp] x W[N][Kp]^T, fused BN+ReLU.
// TDM build: double-buffered LDS, A+B tiles streamed by the Tensor Data Mover,
// pipelined on TENSORcnt (issue next pair, s_wait_tensorcnt 2, compute).
// Fallback: manual cooperative staging.  outF!=nullptr -> final f32 layer.
// ---------------------------------------------------------------------------
__global__ __launch_bounds__(256) void k_fc_wmma(const h16* __restrict__ A,
                                                 const h16* __restrict__ W,
                                                 const float* __restrict__ sc,
                                                 const float* __restrict__ bi,
                                                 h16* __restrict__ outH,
                                                 float* __restrict__ outF,
                                                 int Kp, int ldOut, int doRelu, int nValid) {
  const int tid  = threadIdx.x;
  const int lane = tid & 31;
  const int wave = tid >> 5;
  const int mBase = blockIdx.x * 128;
  const int nBase = blockIdx.y * 32;
  v8f acc0 = {}, acc1 = {};
  const int nStages = Kp >> 6;

#if USE_TDM
  __shared__ __align__(16) h16 As[2][128 * LDK];
  __shared__ __align__(16) h16 Bs[2][32 * LDK];
  const h16* aBase = A + (size_t)mBase * Kp;
  const h16* bBase = W + (size_t)nBase * Kp;
  if (tid == 0) {                               // prologue: stage 0 in flight
    tdm_load_2d((unsigned)(uintptr_t)&As[0][0], aBase,
                (unsigned)Kp, 128u, 64u, 128u, (unsigned)Kp);
    tdm_load_2d((unsigned)(uintptr_t)&Bs[0][0], bBase,
                (unsigned)Kp, 32u, 64u, 32u, (unsigned)Kp);
  }
  for (int kb = 0; kb < nStages; ++kb) {
    const int cur = kb & 1, nxt = cur ^ 1;
    if (tid == 0) {
      if (kb + 1 < nStages) {                   // stream next pair, wait current
        const int ks2 = (kb + 1) << 6;
        tdm_load_2d((unsigned)(uintptr_t)&As[nxt][0], aBase + ks2,
                    (unsigned)(Kp - ks2), 128u, 64u, 128u, (unsigned)Kp);
        tdm_load_2d((unsigned)(uintptr_t)&Bs[nxt][0], bBase + ks2,
                    (unsigned)(Kp - ks2), 32u, 64u, 32u, (unsigned)Kp);
        __builtin_amdgcn_s_wait_tensorcnt((short)2);
      } else {
        __builtin_amdgcn_s_wait_tensorcnt((short)0);
      }
    }
    __syncthreads();
    const int half = lane >> 4, l16 = lane & 15;
    const int kh = half * 8;
    const h16* ap = &As[cur][(wave * 16 + l16) * LDK];
    #pragma unroll
    for (int kc2 = 0; kc2 < 2; ++kc2) {
      const int ksl = kc2 * 32;
      v8h alo = *(const v8h*)(ap + ksl + kh);
      v8h ahi = *(const v8h*)(ap + ksl + 16 + kh);
      v16h a;
      #pragma unroll
      for (int i = 0; i < 8; ++i) { a[i] = alo[i]; a[8 + i] = ahi[i]; }
      #pragma unroll
      for (int ns = 0; ns < 2; ++ns) {
        const h16* bp = &Bs[cur][(ns * 16 + l16) * LDK + ksl + half * 8];
        v8h blo = *(const v8h*)(bp + 0);
        v8h bhi = *(const v8h*)(bp + 16);
        v16h bv;
        #pragma unroll
        for (int i = 0; i < 8; ++i) { bv[i] = blo[i]; bv[8 + i] = bhi[i]; }
        if (ns == 0)
          acc0 = __builtin_amdgcn_wmma_f32_16x16x32_f16(false, a, false, bv,
                                                        (short)0, acc0, false, false);
        else
          acc1 = __builtin_amdgcn_wmma_f32_16x16x32_f16(false, a, false, bv,
                                                        (short)0, acc1, false, false);
      }
    }
    __syncthreads();
  }
#else
  __shared__ __align__(16) h16 As[128 * LDK];
  __shared__ __align__(16) h16 Bs[32 * LDK];
  const int koffA = (tid & 7) * 8;
  const int rA    = tid >> 3;
  const int nB    = tid >> 3;
  const h16* aRow0 = A + (size_t)(mBase + rA) * Kp + koffA;
  const h16* bRow  = W + (size_t)(nBase + nB) * Kp + koffA;
  const size_t aRowStep = (size_t)32 * Kp;
  for (int kb = 0; kb < nStages; ++kb) {
    const int ks = kb << 6;
    #pragma unroll
    for (int i = 0; i < 4; ++i)
      *(v8h*)(As + (rA + 32 * i) * LDK + koffA) = *(const v8h*)(aRow0 + aRowStep * i + ks);
    *(v8h*)(Bs + nB * LDK + koffA) = *(const v8h*)(bRow + ks);
    if (kb + 1 < nStages) __builtin_prefetch(aRow0 + ks + 64, 0, 0);
    __syncthreads();
    const int half = lane >> 4, l16 = lane & 15;
    const int kh = half * 8;
    const h16* ap = As + (wave * 16 + l16) * LDK;
    #pragma unroll
    for (int kc2 = 0; kc2 < 2; ++kc2) {
      const int ksl = kc2 * 32;
      v8h alo = *(const v8h*)(ap + ksl + kh);
      v8h ahi = *(const v8h*)(ap + ksl + 16 + kh);
      v16h a;
      #pragma unroll
      for (int i = 0; i < 8; ++i) { a[i] = alo[i]; a[8 + i] = ahi[i]; }
      #pragma unroll
      for (int ns = 0; ns < 2; ++ns) {
        const h16* bp = Bs + (ns * 16 + l16) * LDK + ksl + half * 8;
        v8h blo = *(const v8h*)(bp + 0);
        v8h bhi = *(const v8h*)(bp + 16);
        v16h bv;
        #pragma unroll
        for (int i = 0; i < 8; ++i) { bv[i] = blo[i]; bv[8 + i] = bhi[i]; }
        if (ns == 0)
          acc0 = __builtin_amdgcn_wmma_f32_16x16x32_f16(false, a, false, bv,
                                                        (short)0, acc0, false, false);
        else
          acc1 = __builtin_amdgcn_wmma_f32_16x16x32_f16(false, a, false, bv,
                                                        (short)0, acc1, false, false);
      }
    }
    __syncthreads();
  }
#endif
  const int half = lane >> 4, l16 = lane & 15;
  #pragma unroll
  for (int ns = 0; ns < 2; ++ns) {
    const int n = nBase + ns * 16 + l16;
    const float s = sc[n], o = bi[n];
    #pragma unroll
    for (int r = 0; r < 8; ++r) {
      int row = mBase + wave * 16 + half * 8 + r;
      float v = (ns == 0 ? acc0[r] : acc1[r]) * s + o;
      if (doRelu) v = v > 0.f ? v : 0.f;
      if (outF) {
        if (n < nValid) outF[(size_t)row * nValid + n] = v;
      } else {
        outH[(size_t)row * ldOut + n] = (h16)v;
      }
    }
  }
}

// ---------------------------------------------------------------------------
// Host launcher
// ---------------------------------------------------------------------------
extern "C" void kernel_launch(void* const* d_in, const int* in_sizes, int n_in,
                              void* d_out, int out_size, void* d_ws, size_t ws_size,
                              hipStream_t stream) {
  (void)in_sizes; (void)n_in; (void)out_size; (void)ws_size;
  const float* cost = (const float*)d_in[0];
  const float* sp   = (const float*)d_in[1];
  const float* gp   = (const float*)d_in[2];
  auto CW = [&](int l, int j) { return (const float*)d_in[3 + l * 6 + j]; };
  auto FW = [&](int l, int j) { return (const float*)d_in[45 + l * 6 + j]; };
  const float* wOut = (const float*)d_in[63];
  const float* bOut = (const float*)d_in[64];

  static const int cinA[7]  = {3, 32, 32, 64, 64, 128, 128};
  static const int coutA[7] = {32, 32, 64, 64, 128, 128, 256};
  static const int lg2A[7]  = {0, 5, 5, 6, 6, 7, 7};

  char* ws = (char*)d_ws;
  size_t off = 0;
  auto take = [&](size_t bytes) -> char* {
    char* p = ws + off;
    off += (bytes + 255) & ~(size_t)255;
    return p;
  };

  h16* convW[7] = {nullptr};
  for (int l = 1; l < 7; ++l)
    convW[l] = (h16*)take((size_t)coutA[l] * 9 * cinA[l] * sizeof(h16));
  float *convS[7], *convBi[7];
  for (int l = 0; l < 7; ++l) {
    convS[l]  = (float*)take(coutA[l] * sizeof(float));
    convBi[l] = (float*)take(coutA[l] * sizeof(float));
  }
  h16* fcW[4];
  fcW[0] = (h16*)take((size_t)512 * KP1 * sizeof(h16));
  fcW[1] = (h16*)take((size_t)256 * 512 * sizeof(h16));
  fcW[2] = (h16*)take((size_t)256 * 256 * sizeof(h16));
  fcW[3] = (h16*)take((size_t)128 * 256 * sizeof(h16));
  static const int fcN[4] = {512, 256, 256, 128};
  float *fcS[4], *fcBi[4];
  for (int j = 0; j < 4; ++j) {
    fcS[j]  = (float*)take(fcN[j] * sizeof(float));
    fcBi[j] = (float*)take(fcN[j] * sizeof(float));
  }
  h16* x0   = (h16*)take((size_t)256 * 120 * 120 * 4 * sizeof(h16));
  h16* bufA = (h16*)take((size_t)256 * 14400 * 32 * sizeof(h16));
  h16* bufB = (h16*)take((size_t)256 * 14400 * 32 * sizeof(h16));
  h16* h0   = (h16*)take((size_t)256 * KP1 * sizeof(h16));
  h16* h1   = (h16*)take((size_t)256 * 512 * sizeof(h16));
  h16* h2   = (h16*)take((size_t)256 * 256 * sizeof(h16));
  h16* h3   = (h16*)take((size_t)256 * 256 * sizeof(h16));

  // ---- parameter prep ----
  for (int l = 1; l < 7; ++l) {
    int n = coutA[l] * 9 * cinA[l];
    k_prep_conv_w<<<(n + 255) / 256, 256, 0, stream>>>(CW(l, 0), convW[l], coutA[l], cinA[l]);
  }
  for (int l = 0; l < 7; ++l)
    k_prep_sb<<<(coutA[l] + 255) / 256, 256, 0, stream>>>(
        CW(l, 1), CW(l, 2), CW(l, 3), CW(l, 4), CW(l, 5), convS[l], convBi[l], coutA[l]);

  { long long n = 512LL * KP1;
    k_prep_fc_w<<<(int)((n + 255) / 256), 256, 0, stream>>>(FW(0, 0), fcW[0], 512, 16388, 512, KP1); }
  { long long n = 256LL * 512;
    k_prep_fc_w<<<(int)((n + 255) / 256), 256, 0, stream>>>(FW(1, 0), fcW[1], 256, 512, 256, 512); }
  { long long n = 256LL * 256;
    k_prep_fc_w<<<(int)((n + 255) / 256), 256, 0, stream>>>(FW(2, 0), fcW[2], 256, 256, 256, 256); }
  { long long n = 128LL * 256;
    k_prep_fc_w<<<(int)((n + 255) / 256), 256, 0, stream>>>(wOut, fcW[3], 100, 256, 128, 256); }
  for (int j = 0; j < 3; ++j)
    k_prep_sb<<<(fcN[j] + 255) / 256, 256, 0, stream>>>(
        FW(j, 1), FW(j, 2), FW(j, 3), FW(j, 4), FW(j, 5), fcS[j], fcBi[j], fcN[j]);
  k_prep_sb_final<<<1, 256, 0, stream>>>(bOut, fcS[3], fcBi[3], 100, 128);

  // ---- forward ----
  k_build_x0<<<(256 * 14400 + 255) / 256, 256, 0, stream>>>(cost, sp, gp, x0);
  { long long n = 256LL * 14400 * 32;
    k_conv1<<<(int)(n / 256), 256, 0, stream>>>(x0, CW(0, 0), convS[0], convBi[0], bufA); }

  auto conv = [&](const h16* in, h16* out, int l, int S) {
    int pos = 256 * S * S;
    dim3 grid((pos + 127) / 128, coutA[l] / 32);
    k_conv_wmma<<<grid, 256, 0, stream>>>(in, out, convW[l], convS[l], convBi[l],
                                          256, S, cinA[l], coutA[l], lg2A[l]);
  };
  auto pool = [&](const h16* in, h16* out, int S, int C) {
    long long n = 256LL * (S / 2) * (S / 2) * C;
    k_maxpool<<<(int)((n + 255) / 256), 256, 0, stream>>>(in, out, 256, S, C);
  };

  conv(bufA, bufB, 1, 120);
  pool(bufB, bufA, 120, 32);
  conv(bufA, bufB, 2, 60);
  conv(bufB, bufA, 3, 60);
  pool(bufA, bufB, 60, 64);
  conv(bufB, bufA, 4, 30);
  conv(bufA, bufB, 5, 30);
  pool(bufB, bufA, 30, 128);
  conv(bufA, bufB, 6, 15);

  k_adpool<<<(256 * 256 * 64) / 256, 256, 0, stream>>>(bufB, h0);
  k_htail<<<(256 * 64) / 256, 256, 0, stream>>>(sp, gp, h0);

  k_fc_wmma<<<dim3(2, 16), 256, 0, stream>>>(h0, fcW[0], fcS[0], fcBi[0], h1, nullptr, KP1, 512, 1, 0);
  k_fc_wmma<<<dim3(2, 8),  256, 0, stream>>>(h1, fcW[1], fcS[1], fcBi[1], h2, nullptr, 512, 256, 1, 0);
  k_fc_wmma<<<dim3(2, 8),  256, 0, stream>>>(h2, fcW[2], fcS[2], fcBi[2], h3, nullptr, 256, 256, 1, 0);
  k_fc_wmma<<<dim3(2, 4),  256, 0, stream>>>(h3, fcW[3], fcS[3], fcBi[3], nullptr, (float*)d_out,
                                             256, 0, 0, 100);
}
